// TransformerModel_4174708211877
// MI455X (gfx1250) — matrix operand order repaired and checked by reading, not compiled, not run on hardware
//
#include <hip/hip_runtime.h>
#include <hip/hip_bf16.h>

// ---------------- types for CDNA5 WMMA / TDM ----------------
typedef __bf16 bf16_t;
typedef __attribute__((ext_vector_type(16))) __bf16 bf16x16;
typedef __attribute__((ext_vector_type(8)))  __bf16 bf16x8;
typedef __attribute__((ext_vector_type(8)))  float  f32x8;
typedef __attribute__((ext_vector_type(4)))  unsigned int u32x4;
typedef __attribute__((ext_vector_type(8)))  int i32x8;
typedef __attribute__((ext_vector_type(4)))  int i32x4;

union FragU {
    bf16x16 v;
    bf16x8  h[2];
    uint4   u[2];
};

__device__ __forceinline__ bf16x16 ld_frag2(const bf16_t* plo, const bf16_t* phi) {
    FragU f;
    f.u[0] = *(const uint4*)plo;   // 8 bf16, 16B
    f.u[1] = *(const uint4*)phi;   // 8 bf16, 16B
    return f.v;
}
__device__ __forceinline__ bf16x16 ld_frag32(const bf16_t* p) {
    return ld_frag2(p, p + 8);     // 16 contiguous bf16, 32B
}

__device__ __forceinline__ f32x8 wmma_bf16(bf16x16 a, bf16x16 b, f32x8 c) {
    // D = A(16x32 bf16) * B(32x16 bf16) + C(16x16 f32)
    return __builtin_amdgcn_wmma_f32_16x16x32_bf16(
        /*neg_a=*/false, a, /*neg_b=*/false, b,
        /*c_mod=*/(short)0, c, /*reuse_a=*/false, /*reuse_b=*/false);
}

__device__ __forceinline__ int rfl(unsigned v) {
    return __builtin_amdgcn_readfirstlane((int)v);
}

// ---------------- Tensor Data Mover: 2D tile -> LDS (with LDS padding) ----
// D# per CDNA5 ISA ch.8: group0 = {ctrl, lds_addr, global_addr, type},
// group1 = {flags/pad, dims, tile dims, stride}. data_size=1 (2-byte bf16).
// pad_interval code 4 => pad after 32 DWORDs (128B row = 64 bf16),
// pad_amount  code 3 => 4 DWORDs (16B = 8 bf16)  -> LDS row stride 72 bf16.
__device__ __forceinline__ void tdm_load_tile_2d(
    unsigned lds_off, const bf16_t* gptr,
    unsigned tensor_d0, unsigned tensor_d1,   // remaining extent from tile start (elems)
    unsigned tile_d0, unsigned tile_d1,       // tile shape (elems)
    unsigned stride0_elems)                   // row stride of tensor (elems)
{
    unsigned long long ga = (unsigned long long)(const void*)gptr;

    u32x4 g0;
    g0[0] = (unsigned)rfl(1u);                                   // count=1, user mode
    g0[1] = (unsigned)rfl(lds_off);                              // lds_addr (bytes)
    g0[2] = (unsigned)rfl((unsigned)(ga & 0xFFFFFFFFu));         // global_addr[31:0]
    g0[3] = (unsigned)rfl((unsigned)((ga >> 32) & 0x1FFFFFFu) |
                          (2u << 30));                           // addr[56:32] | type=2

    // word0: data_size=1 (2B) <<16 | pad_enable<<20 | pad_interval(4)<<22 | pad_amount(3)<<25
    unsigned w0 = (1u << 16) | (1u << 20) | (4u << 22) | (3u << 25);
    i32x8 g1;
    g1[0] = rfl(w0);
    g1[1] = rfl((tensor_d0 & 0xFFFFu) << 16);                    // abar_addr=0 | td0[15:0]
    g1[2] = rfl(((tensor_d0 >> 16) & 0xFFFFu) | ((tensor_d1 & 0xFFFFu) << 16));
    g1[3] = rfl(((tensor_d1 >> 16) & 0xFFFFu) | ((tile_d0 & 0xFFFFu) << 16));
    g1[4] = rfl(tile_d1 & 0xFFFFu);                              // tile_dim1 | tile_dim2=0
    g1[5] = rfl(stride0_elems);                                  // tensor_dim0_stride[31:0]
    g1[6] = rfl(0u);                                             // stride0[47:32] | stride1 lo
    g1[7] = rfl(0u);                                             // stride1 hi

    i32x4 z4; z4[0] = 0; z4[1] = 0; z4[2] = 0; z4[3] = 0;        // groups 2/3 unused (2D)
    i32x8 z8;
#pragma unroll
    for (int i = 0; i < 8; i++) z8[i] = 0;
    // 6-arg toolchain variant: (g0, g1, g2, g3, extra_group, cpol)
    __builtin_amdgcn_tensor_load_to_lds(g0, g1, z4, z4, z8, 0);
}

// ---------------- problem constants ----------------
#define DIMC   4096
#define S_LEN  1023
#define BATCH  4
#define NH     32
#define NKV    8
#define HD     128
#define CS     1024
#define TOTALC 6144            // (32 + 2*8) * 128
#define MROWS  (BATCH * S_LEN) // 4092

// =====================================================================
// 1) Embedding gather: h[row,:] = emb[x[row],:]
// =====================================================================
__global__ void embed_kernel(const int* __restrict__ x,
                             const bf16_t* __restrict__ emb,
                             bf16_t* __restrict__ h) {
    int row = blockIdx.x;
    int tok = x[row];
    const uint4* src = (const uint4*)(emb + (size_t)tok * DIMC);
    uint4*       dst = (uint4*)(h + (size_t)row * DIMC);
    for (int i = threadIdx.x; i < DIMC / 8; i += blockDim.x) dst[i] = src[i];
}

// =====================================================================
// 2) GEMM: C[M,N] = A[M,K] * W[N,K]^T + bias[N]   (bf16 in, f32 acc, bf16 out)
//    128x128 tile per WG, 8 waves, each wave 64x32 = 4x2 of 16x16 WMMA tiles.
//    Tiles staged via the Tensor Data Mover, double-buffered on TENSORcnt.
// =====================================================================
#define BM 128
#define BN 128
#define BK 64
#define LDT (BK + 8)   // padded LDS row stride (144B) -- produced by TDM pad fields

__global__ __launch_bounds__(256) void gemm_bf16_nt(
    const bf16_t* __restrict__ A, const bf16_t* __restrict__ W,
    const bf16_t* __restrict__ bias, bf16_t* __restrict__ C,
    int M, int N, int K)
{
    __shared__ __align__(16) bf16_t As[2][BM * LDT];
    __shared__ __align__(16) bf16_t Bs[2][BN * LDT];

    const int tid  = threadIdx.x;
    const int lane = tid & 31, wave = tid >> 5;
    const int wm   = wave >> 2, wn = wave & 3;     // 2 x 4 wave grid
    const int half = lane >> 4, l15 = lane & 15;
    const int bm   = blockIdx.x * BM, bn = blockIdx.y * BN;

    const f32x8 zero = {0.f, 0.f, 0.f, 0.f, 0.f, 0.f, 0.f, 0.f};
    f32x8 acc[4][2];
#pragma unroll
    for (int i = 0; i < 4; i++)
#pragma unroll
        for (int j = 0; j < 2; j++) acc[i][j] = zero;

    // ---- TDM prologue: stage 0 into buffer 0 (wave 0 drives the DMA) ----
    if (wave == 0) {
        tdm_load_tile_2d((unsigned)(size_t)&As[0][0], A + (size_t)bm * K,
                         (unsigned)K, (unsigned)(M - bm), BK, BM, (unsigned)K);
        tdm_load_tile_2d((unsigned)(size_t)&Bs[0][0], W + (size_t)bn * K,
                         (unsigned)K, (unsigned)(N - bn), BK, BN, (unsigned)K);
    }

    int stage = 0;
    for (int kb = 0; kb < K; kb += BK, stage ^= 1) {
        if (wave == 0) {
            if (kb + BK < K) {
                // prefetch next stage into the other buffer, then wait for current
                tdm_load_tile_2d((unsigned)(size_t)&As[stage ^ 1][0],
                                 A + (size_t)bm * K + kb + BK,
                                 (unsigned)(K - kb - BK), (unsigned)(M - bm),
                                 BK, BM, (unsigned)K);
                tdm_load_tile_2d((unsigned)(size_t)&Bs[stage ^ 1][0],
                                 W + (size_t)bn * K + kb + BK,
                                 (unsigned)(K - kb - BK), (unsigned)(N - bn),
                                 BK, BN, (unsigned)K);
                __builtin_amdgcn_s_wait_tensorcnt(2);  // in-order: current stage landed
            } else {
                __builtin_amdgcn_s_wait_tensorcnt(0);
            }
        }
        __syncthreads();   // current stage visible to all waves

        const bf16_t* as = &As[stage][0];
        const bf16_t* bs = &Bs[stage][0];
#pragma unroll
        for (int ks = 0; ks < BK; ks += 32) {
            bf16x16 af[4], bfm[2];
#pragma unroll
            for (int tm = 0; tm < 4; tm++) {
                const bf16_t* p = as + (wm * 64 + tm * 16 + l15) * LDT + ks + half * 8;
                af[tm] = ld_frag2(p, p + 16);   // K {0-7,16-23} / {8-15,24-31}
            }
#pragma unroll
            for (int tn = 0; tn < 2; tn++) {
                const bf16_t* p = bs + (wn * 32 + tn * 16 + l15) * LDT + ks + half * 8;
                bfm[tn] = ld_frag2(p, p + 16);  // column = weight row, K {0-7,16-23} / {8-15,24-31}
            }
#pragma unroll
            for (int tm = 0; tm < 4; tm++)
#pragma unroll
                for (int tn = 0; tn < 2; tn++)
                    acc[tm][tn] = wmma_bf16(af[tm], bfm[tn], acc[tm][tn]);
        }
        __syncthreads();   // reads done before TDM overwrites this buffer
    }

    // epilogue: bias + bf16 store (D layout: VGPR r -> row r + 8*half, lane -> col)
#pragma unroll
    for (int tn = 0; tn < 2; tn++) {
        int col  = bn + wn * 32 + tn * 16 + l15;
        float bv = (float)bias[col];
#pragma unroll
        for (int tm = 0; tm < 4; tm++)
#pragma unroll
            for (int r = 0; r < 8; r++) {
                int row = bm + wm * 64 + tm * 16 + r + 8 * half;
                if (row < M) C[(size_t)row * N + col] = (bf16_t)(acc[tm][tn][r] + bv);
            }
    }
}

// =====================================================================
// 3) RoPE in-place on q (heads 0..31) and k (heads 32..39) of qkv
// =====================================================================
__global__ void rope_kernel(bf16_t* __restrict__ qkv, const float* __restrict__ freqs) {
    int bs = blockIdx.x;                 // b*S + s
    int s  = bs % S_LEN;
    const float* f = freqs + (size_t)s * HD;   // [64][2] f32
    bf16_t* row = qkv + (size_t)bs * TOTALC;
    for (int p = threadIdx.x; p < (NH + NKV) * 64; p += blockDim.x) {
        int head = p >> 6, i = p & 63;
        int base = head * HD + i * 2;     // k heads land at 4096..5119
        float x0 = (float)row[base], x1 = (float)row[base + 1];
        float f0 = f[i * 2], f1 = f[i * 2 + 1];
        row[base]     = (bf16_t)(x0 * f0 - x1 * f1);
        row[base + 1] = (bf16_t)(x1 * f0 + x0 * f1);
    }
}

// =====================================================================
// 4) KV cache scatter: rotated k and v -> [B,NKV,CS,HD] buffers
// =====================================================================
__global__ void kv_scatter(const bf16_t* __restrict__ qkv, const int* __restrict__ input_pos,
                           bf16_t* __restrict__ kf, bf16_t* __restrict__ vf) {
    int bs = blockIdx.x;
    int b = bs / S_LEN, s = bs % S_LEN;
    int pos = input_pos[s];
    int t = threadIdx.x;                  // 128 threads, one 16B chunk each
    int kvh = t >> 4;
    int d   = (t & 15) * 8;
    size_t dst = ((((size_t)b * NKV + kvh) * CS + pos) * HD + d);
    const bf16_t* src = qkv + (size_t)bs * TOTALC;
    *(uint4*)(kf + dst) = *(const uint4*)(src + DIMC + t * 8);
    *(uint4*)(vf + dst) = *(const uint4*)(src + DIMC + NKV * HD + t * 8);
}

// =====================================================================
// 5) V transpose: [B*NKV][CS][HD] -> [B*NKV][HD][CS]  (for PV B-fragments)
// =====================================================================
__global__ void v_transpose(const bf16_t* __restrict__ vf, bf16_t* __restrict__ vT) {
    int bh  = blockIdx.y;
    int idx = blockIdx.x * blockDim.x + threadIdx.x;   // 0 .. CS*HD-1
    int s = idx >> 7, d = idx & 127;
    vT[(size_t)bh * HD * CS + (size_t)d * CS + s] =
        vf[(size_t)bh * CS * HD + (size_t)s * HD + d];
}

// =====================================================================
// 6) Flash attention (GQA, non-causal over full 1024-entry cache)
//    grid (qtiles=8, heads=32, batch=4), 8 waves, 16 q rows per wave
// =====================================================================
#define PST 40   // P LDS row stride (80B, 16B-aligned)

__global__ __launch_bounds__(256) void attn_kernel(
    const bf16_t* __restrict__ qkv,   // rotated q lives in first 4096 cols
    const bf16_t* __restrict__ kf,    // [B,NKV,CS,HD]
    const bf16_t* __restrict__ vT,    // [B,NKV,HD,CS]
    bf16_t* __restrict__ y)           // [B,S,DIM]
{
    __shared__ __align__(16) bf16_t p_lds[8][16 * PST];

    const int lane = threadIdx.x & 31, wave = threadIdx.x >> 5;
    const int half = lane >> 4, l15 = lane & 15;
    const int qt = blockIdx.x, h = blockIdx.y, b = blockIdx.z;
    const int kvh = h >> 2;

    const int qrow = qt * 128 + wave * 16 + l15;
    const int qr   = qrow < S_LEN ? qrow : S_LEN - 1;
    const bf16_t* qp = qkv + (size_t)(b * S_LEN + qr) * TOTALC + h * HD;

    bf16x16 aq[4];
#pragma unroll
    for (int kc = 0; kc < 4; kc++)
        aq[kc] = ld_frag2(qp + kc * 32 + half * 8, qp + kc * 32 + 16 + half * 8);

    const bf16_t* kbase = kf + ((size_t)b * NKV + kvh) * CS * HD;
    const bf16_t* vbase = vT + ((size_t)b * NKV + kvh) * HD * CS;
    bf16_t* pl = &p_lds[wave][0];

    const f32x8 zero = {0.f, 0.f, 0.f, 0.f, 0.f, 0.f, 0.f, 0.f};
    f32x8 acc[8];
#pragma unroll
    for (int t = 0; t < 8; t++) acc[t] = zero;
    float m_i[8], l_i[8];
#pragma unroll
    for (int r = 0; r < 8; r++) { m_i[r] = -1e30f; l_i[r] = 0.f; }

    const float scale = 0.08838834764831845f;   // 1/sqrt(128)

    for (int j = 0; j < CS; j += 32) {
        // ---- scores: Q[16x128] x K^T[128x32] -> two 16x16 f32 tiles ----
        f32x8 c0 = zero, c1 = zero;
#pragma unroll
        for (int kc = 0; kc < 4; kc++) {
            bf16x16 b0 = ld_frag2(kbase + (size_t)(j + l15) * HD + kc * 32 + half * 8, kbase + (size_t)(j + l15) * HD + kc * 32 + 16 + half * 8);
            bf16x16 b1 = ld_frag2(kbase + (size_t)(j + 16 + l15) * HD + kc * 32 + half * 8, kbase + (size_t)(j + 16 + l15) * HD + kc * 32 + 16 + half * 8);
            c0 = wmma_bf16(aq[kc], b0, c0);
            c1 = wmma_bf16(aq[kc], b1, c1);
        }

        // ---- online softmax (per-row over 16-lane halves) ----
        float p0[8], p1[8], alpha[8];
#pragma unroll
        for (int r = 0; r < 8; r++) {
            float s0 = c0[r] * scale, s1 = c1[r] * scale;
            float mx = fmaxf(s0, s1);
#pragma unroll
            for (int m = 1; m <= 8; m <<= 1) mx = fmaxf(mx, __shfl_xor(mx, m, 32));
            float mn = fmaxf(m_i[r], mx);
            alpha[r] = __expf(m_i[r] - mn);
            p0[r] = __expf(s0 - mn);
            p1[r] = __expf(s1 - mn);
            float rs = p0[r] + p1[r];
#pragma unroll
            for (int m = 1; m <= 8; m <<= 1) rs += __shfl_xor(rs, m, 32);
            l_i[r] = l_i[r] * alpha[r] + rs;
            m_i[r] = mn;
        }
#pragma unroll
        for (int t = 0; t < 8; t++)
#pragma unroll
            for (int r = 0; r < 8; r++) acc[t][r] *= alpha[r];

        // ---- re-layout P: score layout -> A-fragment layout via wave LDS ----
#pragma unroll
        for (int r = 0; r < 8; r++) {
            int row = r + 8 * half;
            pl[row * PST + l15]      = (bf16_t)p0[r];
            pl[row * PST + 16 + l15] = (bf16_t)p1[r];
        }
        asm volatile("s_wait_dscnt 0" ::: "memory");   // wave-local LDS RAW
        bf16x16 ap = ld_frag2(pl + l15 * PST + half * 8,
                              pl + l15 * PST + 16 + half * 8);

        // ---- PV: P[16x32] x V[32x128] accumulated into 8 d-tiles ----
#pragma unroll
        for (int t = 0; t < 8; t++) {
            bf16x16 bv = ld_frag2(vbase + (size_t)(t * 16 + l15) * CS + j + half * 8, vbase + (size_t)(t * 16 + l15) * CS + j + 16 + half * 8);
            acc[t] = wmma_bf16(ap, bv, acc[t]);
        }
    }

    // ---- normalize + store y[b, row, h*128 + d] ----
#pragma unroll
    for (int r = 0; r < 8; r++) {
        int row = qt * 128 + wave * 16 + r + 8 * half;
        if (row < S_LEN) {
            float inv = 1.0f / l_i[r];
            bf16_t* yp = y + (size_t)(b * S_LEN + row) * DIMC + h * HD;
#pragma unroll
            for (int t = 0; t < 8; t++)
                yp[t * 16 + l15] = (bf16_t)(acc[t][r] * inv);
        }
    }
}

// =====================================================================
// launcher
// =====================================================================
extern "C" void kernel_launch(void* const* d_in, const int* in_sizes, int n_in,
                              void* d_out, int out_size, void* d_ws, size_t ws_size,
                              hipStream_t stream) {
    const int*    x         = (const int*)d_in[0];
    const float*  freqs_cis = (const float*)d_in[1];
    // d_in[2] = mask (unused by the reference SDPA)
    const int*    input_pos = (const int*)d_in[3];
    const bf16_t* k_cache   = (const bf16_t*)d_in[4];
    const bf16_t* v_cache   = (const bf16_t*)d_in[5];
    const bf16_t* emb       = (const bf16_t*)d_in[6];
    const bf16_t* wqkv_w    = (const bf16_t*)d_in[7];
    const bf16_t* wqkv_b    = (const bf16_t*)d_in[8];
    const bf16_t* wo_w      = (const bf16_t*)d_in[9];
    const bf16_t* wo_b      = (const bf16_t*)d_in[10];
    bf16_t*       out       = (bf16_t*)d_out;

    const size_t H_ELEMS   = (size_t)MROWS * DIMC;          // 16,760,832
    const size_t QKV_ELEMS = (size_t)MROWS * TOTALC;        // 25,141,248
    const size_t KV_ELEMS  = (size_t)BATCH * NKV * CS * HD; // 4,194,304

    bf16_t* ws  = (bf16_t*)d_ws;
    bf16_t* h   = ws;                   // reused as attention output y later
    bf16_t* qkv = h + H_ELEMS;
    bf16_t* kf  = qkv + QKV_ELEMS;
    bf16_t* vf  = kf + KV_ELEMS;
    bf16_t* vT  = vf + KV_ELEMS;

    // 1) embedding gather
    embed_kernel<<<dim3(MROWS), dim3(256), 0, stream>>>(x, emb, h);

    // 2) fused QKV projection: [4092,6144] = h[4092,4096] * wqkv^T + b
    gemm_bf16_nt<<<dim3(32, 48), dim3(256), 0, stream>>>(
        h, wqkv_w, wqkv_b, qkv, MROWS, TOTALC, DIMC);

    // 3) RoPE on q and k (in place)
    rope_kernel<<<dim3(MROWS), dim3(256), 0, stream>>>(qkv, freqs_cis);

    // 4) cache init (preserves zero row at pos 1023) + scatter update
    (void)hipMemcpyAsync(kf, k_cache, KV_ELEMS * sizeof(bf16_t), hipMemcpyDeviceToDevice, stream);
    (void)hipMemcpyAsync(vf, v_cache, KV_ELEMS * sizeof(bf16_t), hipMemcpyDeviceToDevice, stream);
    kv_scatter<<<dim3(MROWS), dim3(128), 0, stream>>>(qkv, input_pos, kf, vf);

    // 5) V transpose for PV B-fragments
    v_transpose<<<dim3(512, BATCH * NKV), dim3(256), 0, stream>>>(vf, vT);

    // 6) flash attention -> y (reuses h buffer)
    attn_kernel<<<dim3(8, NH, BATCH), dim3(256), 0, stream>>>(qkv, kf, vT, h);

    // 7) output projection: out = y * wo^T + b
    gemm_bf16_nt<<<dim3(32, 32), dim3(256), 0, stream>>>(
        h, wo_w, wo_b, out, MROWS, DIMC, DIMC);
}
